// GatedDeltaNet_30288109371567
// MI455X (gfx1250) — matrix operand order repaired and checked by reading, not compiled, not run on hardware
//
#include <hip/hip_runtime.h>
#include <cstdint>

typedef __attribute__((ext_vector_type(16))) _Float16 v16h;
typedef __attribute__((ext_vector_type(8)))  _Float16 v8h;
typedef __attribute__((ext_vector_type(8)))  float    v8f;

#define D64    64
#define TILE_T 128
#define HROWS  130      // valid halo rows (t0-1 .. t0+128)
#define MPAD   144      // rows padded to 9 WMMA m-tiles
#define SH16   72       // f16 LDS row stride (144B -> 4-bank shift per row)
#define SF32   68       // f32 LDS row stride (272B -> 4-bank shift per row)
#define EPSV   1e-8f

// shared memory layout (bytes); all offsets 16B aligned
#define OFF_H     0
#define OFF_WQ    (OFF_H  + MPAD * SH16 * 2)     // 20736
#define OFF_WK    (OFF_WQ + 64 * SH16 * 2)       // +9216
#define OFF_WV    (OFF_WK + 64 * SH16 * 2)
#define OFF_Q     (OFF_WV + 64 * SH16 * 2)       // 48384
#define OFF_K     (OFF_Q  + MPAD * SH16 * 2)
#define OFF_V     (OFF_K  + MPAD * SH16 * 2)
#define OFF_DELTA (OFF_V  + MPAD * SH16 * 2)     // 110592
#define SMEM_BYTES (OFF_DELTA + TILE_T * SF32 * 4) // 145408

static_assert(SMEM_BYTES <= 160 * 1024, "fit 2 WGs per 320KB WGP");

__device__ __forceinline__ float wred(float v) {
#pragma unroll
  for (int off = 16; off > 0; off >>= 1) v += __shfl_xor(v, off, 32);
  return v;
}
__device__ __forceinline__ float sigm(float v) { return 1.f / (1.f + __expf(-v)); }

// A-matrix 16-bit 16x32 fragment: lanes 0-15 hold K kb..kb+7 (v0-3) and
// kb+16..kb+23 (v4-7); lanes 16-31 are +8 within each group.
__device__ __forceinline__ v16h load_a_frag(const _Float16* arow, int kbase, int half) {
  v8h lo = *(const v8h*)(arow + kbase + half * 8);
  v8h hi = *(const v8h*)(arow + kbase + 16 + half * 8);
  v16h a;
#pragma unroll
  for (int i = 0; i < 8; ++i) { a[i] = lo[i]; a[i + 8] = hi[i]; }
  return a;
}
// B-matrix 16-bit 32x16 fragment: lanes 0-15 hold K kb..kb+7 and kb+16..kb+23,
// lanes 16-31 are +8 within each group (same as A).
__device__ __forceinline__ v16h load_b_frag(const _Float16* brow, int kbase, int half) {
  v8h lo = *(const v8h*)(brow + kbase + half * 8);
  v8h hi = *(const v8h*)(brow + kbase + 16 + half * 8);
  v16h b;
#pragma unroll
  for (int i = 0; i < 8; ++i) { b[i] = lo[i]; b[i + 8] = hi[i]; }
  return b;
}
// 16x16 output tile with K=64 = two chained v_wmma_f32_16x16x32_f16
__device__ __forceinline__ v8f wmma_2k(v16h a0, v16h a1, const _Float16* sB, int nt, int lane) {
  int mrow = lane & 15, half = lane >> 4;
  const _Float16* brow = sB + (nt * 16 + mrow) * SH16;
  v16h b0 = load_b_frag(brow, 0, half);
  v16h b1 = load_b_frag(brow, 32, half);
  v8f acc = {};
  acc = __builtin_amdgcn_wmma_f32_16x16x32_f16(false, a0, false, b0, (short)0, acc, false, false);
  acc = __builtin_amdgcn_wmma_f32_16x16x32_f16(false, a1, false, b1, (short)0, acc, false, false);
  return acc;
}

extern "C" __global__ __launch_bounds__(256)
void gdn_fused(const float* __restrict__ x,
               const float* __restrict__ norm_g,
               const float* __restrict__ qw, const float* __restrict__ qb,
               const float* __restrict__ kw, const float* __restrict__ kbias,
               const float* __restrict__ vw, const float* __restrict__ vb,
               const float* __restrict__ qcw, const float* __restrict__ qcb,
               const float* __restrict__ kcw, const float* __restrict__ kcb,
               const float* __restrict__ vcw, const float* __restrict__ vcb,
               const float* __restrict__ aw, const float* __restrict__ abias,
               const float* __restrict__ bw, const float* __restrict__ bbias,
               const float* __restrict__ png,
               const float* __restrict__ pw, const float* __restrict__ pb,
               float* __restrict__ out, int T)
{
  extern __shared__ __align__(16) char smem[];
  _Float16* sH  = (_Float16*)(smem + OFF_H);
  _Float16* sWq = (_Float16*)(smem + OFF_WQ);
  _Float16* sWk = (_Float16*)(smem + OFF_WK);
  _Float16* sWv = (_Float16*)(smem + OFF_WV);
  _Float16* sQ  = (_Float16*)(smem + OFF_Q);
  _Float16* sK  = (_Float16*)(smem + OFF_K);
  _Float16* sV  = (_Float16*)(smem + OFF_V);
  float*    sDelta = (float*)(smem + OFF_DELTA);

  const int tilesPerSeq = T / TILE_T;
  const int bIdx = blockIdx.x / tilesPerSeq;
  const int t0   = (blockIdx.x % tilesPerSeq) * TILE_T;
  const int tid  = threadIdx.x;
  const int wave = tid >> 5;
  const int lane = tid & 31;
  const int mrow = lane & 15;
  const int half = lane >> 4;
  const size_t seqBase = (size_t)bIdx * (size_t)T * D64;

  // load q/k/v projection weights to LDS as f16 ([out][in] row-major)
  for (int idx = tid; idx < 64 * 64; idx += 256) {
    int n = idx >> 6, k2 = idx & 63;
    sWq[n * SH16 + k2] = (_Float16)qw[idx];
    sWk[n * SH16 + k2] = (_Float16)kw[idx];
    sWv[n * SH16 + k2] = (_Float16)vw[idx];
  }

  // ---- phase 1: h = zc_rmsnorm(x, norm_g) over halo rows -> f16 sH ----
  for (int hr = wave; hr < MPAD; hr += 8) {
    int t = t0 - 1 + hr;
    bool valid = (hr < HROWS) && (t >= 0) && (t < T);
    float v0 = 0.f, v1 = 0.f;
    if (valid) {
      const float* px = x + seqBase + (size_t)t * D64;
      v0 = px[lane]; v1 = px[lane + 32];
    }
    float mean = wred(v0 + v1) * (1.f / 64.f);
    float c0 = v0 - mean, c1 = v1 - mean;
    float rs = rsqrtf(wred(c0 * c0 + c1 * c1) * (1.f / 64.f) + EPSV);
    _Float16* row = sH + hr * SH16;
    if (valid) {
      row[lane]      = (_Float16)(c0 * rs * norm_g[lane]);
      row[lane + 32] = (_Float16)(c1 * rs * norm_g[lane + 32]);
    } else {
      row[lane] = (_Float16)0.f;
      row[lane + 32] = (_Float16)0.f;
    }
  }
  __syncthreads();

  // ---- phase 2: q/k/v linears via WMMA (9 m-tiles x 4 n-tiles each) ----
  for (int tile = wave; tile < 9 * 4; tile += 8) {
    int mt = tile >> 2, nt = tile & 3;
    const _Float16* arow = sH + (mt * 16 + mrow) * SH16;
    v16h a0 = load_a_frag(arow, 0, half);
    v16h a1 = load_a_frag(arow, 32, half);
    v8f aq = wmma_2k(a0, a1, sWq, nt, lane);
    v8f ak = wmma_2k(a0, a1, sWk, nt, lane);
    v8f av = wmma_2k(a0, a1, sWv, nt, lane);
    int col = nt * 16 + mrow;
    int m0  = mt * 16 + half * 8;
    float biasq = qb[col], biask = kbias[col], biasv = vb[col];
#pragma unroll
    for (int j = 0; j < 8; ++j) {
      sQ[(m0 + j) * SH16 + col] = (_Float16)(aq[j] + biasq);
      sK[(m0 + j) * SH16 + col] = (_Float16)(ak[j] + biask);
      sV[(m0 + j) * SH16 + col] = (_Float16)(av[j] + biasv);
    }
  }
  __syncthreads();

  // ---- phase 3: dwconv3 (zero-pad on conv input) + sigmoid + l2norm + delta ----
  {
    int c0 = lane, c1 = lane + 32;
    float wq0[3], wq1[3], wk0[3], wk1[3], wv0[3], wv1[3];
#pragma unroll
    for (int j = 0; j < 3; ++j) {
      wq0[j] = qcw[c0 * 3 + j]; wq1[j] = qcw[c1 * 3 + j];
      wk0[j] = kcw[c0 * 3 + j]; wk1[j] = kcw[c1 * 3 + j];
      wv0[j] = vcw[c0 * 3 + j]; wv1[j] = vcw[c1 * 3 + j];
    }
    float bq0 = qcb[c0], bq1 = qcb[c1];
    float bk0 = kcb[c0], bk1 = kcb[c1];
    float bv0 = vcb[c0], bv1 = vcb[c1];
    for (int r = wave; r < TILE_T; r += 8) {
      float aq0 = bq0, aq1 = bq1, ak0 = bk0, ak1 = bk1, av0 = bv0, av1 = bv1;
#pragma unroll
      for (int j = 0; j < 3; ++j) {
        int t = t0 + r - 1 + j;
        float m = (t >= 0 && t < T) ? 1.f : 0.f;
        int ro = (r + j) * SH16;   // halo row index: hr = r + j
        aq0 += m * wq0[j] * (float)sQ[ro + c0];
        aq1 += m * wq1[j] * (float)sQ[ro + c1];
        ak0 += m * wk0[j] * (float)sK[ro + c0];
        ak1 += m * wk1[j] * (float)sK[ro + c1];
        av0 += m * wv0[j] * (float)sV[ro + c0];
        av1 += m * wv1[j] * (float)sV[ro + c1];
      }
      float q0 = sigm(aq0), q1 = sigm(aq1);
      float k0 = sigm(ak0), k1 = sigm(ak1);
      float v0 = sigm(av0), v1 = sigm(av1);
      float qrs = rsqrtf(wred(q0 * q0 + q1 * q1) + EPSV);
      float krs = rsqrtf(wred(k0 * k0 + k1 * k1) + EPSV);
      sDelta[r * SF32 + c0] = (q0 * qrs) * (k0 * krs) * v0;
      sDelta[r * SF32 + c1] = (q1 * qrs) * (k1 * krs) * v1;
    }
  }

  // reload weights (aw/bw/pw) and x (interior) as f16 A-matrix into sH
  for (int idx = tid; idx < 64 * 64; idx += 256) {
    int n = idx >> 6, k2 = idx & 63;
    sWq[n * SH16 + k2] = (_Float16)aw[idx];
    sWk[n * SH16 + k2] = (_Float16)bw[idx];
    sWv[n * SH16 + k2] = (_Float16)pw[idx];
  }
  for (int idx = tid; idx < TILE_T * D64; idx += 256) {
    int r = idx >> 6, c = idx & 63;
    sH[r * SH16 + c] = (_Float16)x[seqBase + (size_t)(t0 + r) * D64 + c];
  }
  __syncthreads();

  // ---- phase 4: delta = tanh(x@aw^T+ab) * delta + (x@bw^T+bb) ----
  for (int tile = wave; tile < 8 * 4; tile += 8) {
    int mt = tile >> 2, nt = tile & 3;
    const _Float16* arow = sH + (mt * 16 + mrow) * SH16;
    v16h a0 = load_a_frag(arow, 0, half);
    v16h a1 = load_a_frag(arow, 32, half);
    v8f aa  = wmma_2k(a0, a1, sWq, nt, lane);
    v8f abf = wmma_2k(a0, a1, sWk, nt, lane);
    int col = nt * 16 + mrow;
    int m0  = mt * 16 + half * 8;
    float biasa = abias[col], biasb = bbias[col];
#pragma unroll
    for (int j = 0; j < 8; ++j) {
      float* pd = &sDelta[(m0 + j) * SF32 + col];
      float dv = *pd;
      *pd = tanhf(aa[j] + biasa) * dv + (abf[j] + biasb);
    }
  }
  __syncthreads();

  // ---- phase 5: zc_rmsnorm(delta, post_norm_g) -> f16 into sH ----
  for (int r = wave; r < TILE_T; r += 8) {
    float d0 = sDelta[r * SF32 + lane];
    float d1 = sDelta[r * SF32 + lane + 32];
    float mean = wred(d0 + d1) * (1.f / 64.f);
    float c0 = d0 - mean, c1 = d1 - mean;
    float rs = rsqrtf(wred(c0 * c0 + c1 * c1) * (1.f / 64.f) + EPSV);
    sH[r * SH16 + lane]      = (_Float16)(c0 * rs * png[lane]);
    sH[r * SH16 + lane + 32] = (_Float16)(c1 * rs * png[lane + 32]);
  }
  __syncthreads();

  // ---- phase 6: dhat = dnorm @ pw^T + pb; out = x + sigmoid(silu(dhat))*dhat ----
  for (int tile = wave; tile < 8 * 4; tile += 8) {
    int mt = tile >> 2, nt = tile & 3;
    const _Float16* arow = sH + (mt * 16 + mrow) * SH16;
    v16h a0 = load_a_frag(arow, 0, half);
    v16h a1 = load_a_frag(arow, 32, half);
    v8f acc = wmma_2k(a0, a1, sWv, nt, lane);
    int col = nt * 16 + mrow;
    int m0  = mt * 16 + half * 8;
    float biasp = pb[col];
#pragma unroll
    for (int j = 0; j < 8; ++j) {
      float dh   = acc[j] + biasp;
      float silu = dh * sigm(dh);
      float gate = sigm(silu);
      size_t gi = seqBase + (size_t)(t0 + m0 + j) * D64 + col;
      out[gi] = x[gi] + gate * dh;
    }
  }
}

extern "C" void kernel_launch(void* const* d_in, const int* in_sizes, int n_in,
                              void* d_out, int out_size, void* d_ws, size_t ws_size,
                              hipStream_t stream) {
  const float* x    = (const float*)d_in[0];
  const float* ng   = (const float*)d_in[1];
  const float* qw   = (const float*)d_in[2];
  const float* qb   = (const float*)d_in[3];
  const float* kw   = (const float*)d_in[4];
  const float* kbv  = (const float*)d_in[5];
  const float* vw   = (const float*)d_in[6];
  const float* vb   = (const float*)d_in[7];
  const float* qcw  = (const float*)d_in[8];
  const float* qcb  = (const float*)d_in[9];
  const float* kcw  = (const float*)d_in[10];
  const float* kcb  = (const float*)d_in[11];
  const float* vcw  = (const float*)d_in[12];
  const float* vcb  = (const float*)d_in[13];
  const float* aw   = (const float*)d_in[14];
  const float* abv  = (const float*)d_in[15];
  const float* bw   = (const float*)d_in[16];
  const float* bbv  = (const float*)d_in[17];
  const float* png  = (const float*)d_in[18];
  const float* pw   = (const float*)d_in[19];
  const float* pb   = (const float*)d_in[20];
  float* out = (float*)d_out;

  const int T = 4096;
  const int B = in_sizes[0] / (T * D64);
  dim3 grid(B * (T / TILE_T));
  gdn_fused<<<grid, dim3(256), SMEM_BYTES, stream>>>(
      x, ng, qw, qb, kw, kbv, vw, vb, qcw, qcb, kcw, kcb, vcw, vcb,
      aw, abv, bw, bbv, png, pw, pb, out, T);
}
